// PointConvDensitySetPropagation_58737972740842
// MI455X (gfx1250) — matrix operand order repaired and checked by reading, not compiled, not run on hardware
//
#include <hip/hip_runtime.h>

// ---------------- problem constants (match reference) ----------------
#define BB   2
#define NN   4096
#define SS   1024
#define KK   16
#define DD1  128
#define DD2  256
#define CIN  259                 // 3 + D2
#define CK   4144                // CIN * KK
#define CKP  4160                // padded to multiple of 32 for WMMA K-loop
#define BN_TOT (BB * NN)         // 8192 points
#define KN   (KK * NN)           // 65536 positions per batch for the small MLPs
#define EPS_BN 1e-5f

// ---------------- WMMA types ----------------
typedef __bf16 v16bf __attribute__((ext_vector_type(16)));
typedef float  v8f   __attribute__((ext_vector_type(8)));
typedef unsigned int v8u __attribute__((ext_vector_type(8)));

union BF16x16 { v8u u; v16bf b; };

__device__ __forceinline__ unsigned short f2bf(float x) {
  unsigned int u = __builtin_bit_cast(unsigned int, x);
  unsigned int r = (u + 0x7FFFu + ((u >> 16) & 1u)) >> 16;   // round-to-nearest-even
  return (unsigned short)r;
}

// ============================================================
// 1) transpose points2 [B,D2,S] -> p2t [B,S,D2]
// ============================================================
__global__ void k_transpose_p2(const float* __restrict__ p2, float* __restrict__ p2t) {
  long e = (long)blockIdx.x * 256 + threadIdx.x;
  long total = (long)BB * SS * DD2;
  if (e >= total) return;
  int b = (int)(e / ((long)SS * DD2));
  int rem = (int)(e % ((long)SS * DD2));
  int s = rem / DD2, c = rem % DD2;
  p2t[e] = p2[((long)b * DD2 + c) * SS + s];
}

// ============================================================
// 2) 3-NN of xyz1 against xyz2 (per-point), weights + indices
// ============================================================
__global__ void k_3nn(const float* __restrict__ xyz1, const float* __restrict__ xyz2,
                      int* __restrict__ idx3, float* __restrict__ w3) {
  __shared__ float sx[SS], sy[SS], sz[SS];
  int blocksPerBatch = NN / 256;
  int b = blockIdx.x / blocksPerBatch;
  int row = (blockIdx.x % blocksPerBatch) * 256 + threadIdx.x;
  const float* x2 = xyz2 + (long)b * 3 * SS;
  for (int i = threadIdx.x; i < SS; i += 256) {
    sx[i] = x2[i]; sy[i] = x2[SS + i]; sz[i] = x2[2 * SS + i];
  }
  __syncthreads();
  const float* x1 = xyz1 + (long)b * 3 * NN;
  float xi = x1[row], yi = x1[NN + row], zi = x1[2 * NN + row];
  float d0 = 3.4e38f, d1 = 3.4e38f, d2 = 3.4e38f;
  int i0 = 0, i1 = 0, i2 = 0;
  for (int j = 0; j < SS; ++j) {
    float dx = xi - sx[j], dy = yi - sy[j], dz = zi - sz[j];
    float dist = dx * dx + dy * dy + dz * dz;
    if (dist < d2) {
      if (dist < d0)      { d2 = d1; i2 = i1; d1 = d0; i1 = i0; d0 = dist; i0 = j; }
      else if (dist < d1) { d2 = d1; i2 = i1; d1 = dist; i1 = j; }
      else                { d2 = dist; i2 = j; }
    }
  }
  float a0 = 1.0f / (d0 + 1e-8f), a1 = 1.0f / (d1 + 1e-8f), a2 = 1.0f / (d2 + 1e-8f);
  float inv = 1.0f / (a0 + a1 + a2);
  long base = ((long)b * NN + row) * 3;
  idx3[base] = i0; idx3[base + 1] = i1; idx3[base + 2] = i2;
  w3[base] = a0 * inv; w3[base + 1] = a1 * inv; w3[base + 2] = a2 * inv;
}

// ============================================================
// 3) gather+blend: interp[b,n,c] = sum_i w_i * p2t[b,idx_i,c]
//    one block (256 thr) per point; c == threadIdx.x
// ============================================================
__global__ void k_interp(const float* __restrict__ p2t, const int* __restrict__ idx3,
                         const float* __restrict__ w3, float* __restrict__ interp) {
  int p = blockIdx.x;                 // 0..BN_TOT-1
  int b = p / NN;
  int t = threadIdx.x;
  long base = (long)p * 3;
  int j0 = idx3[base], j1 = idx3[base + 1], j2 = idx3[base + 2];
  float a0 = w3[base], a1 = w3[base + 1], a2 = w3[base + 2];
  const float* r0 = p2t + ((long)b * SS + j0) * DD2;
  const float* r1 = p2t + ((long)b * SS + j1) * DD2;
  const float* r2 = p2t + ((long)b * SS + j2) * DD2;
  interp[(long)p * DD2 + t] = a0 * r0[t] + a1 * r1[t] + a2 * r2[t];
}

// ============================================================
// 4) N x N brute-force kNN (top-16) + gaussian density + gxyz_norm
//    thread per row; LDS-tiled candidate points; register bubble-insert
// ============================================================
__global__ void k_knn_density(const float* __restrict__ xyz1, int* __restrict__ gidx,
                              float* __restrict__ invd, float* __restrict__ gxyzn) {
  __shared__ float tx[256], ty[256], tz[256];
  int blocksPerBatch = NN / 256;
  int b = blockIdx.x / blocksPerBatch;
  int row = (blockIdx.x % blocksPerBatch) * 256 + threadIdx.x;
  const float* xb = xyz1 + (long)b * 3 * NN;
  float xi = xb[row], yi = xb[NN + row], zi = xb[2 * NN + row];
  float d[KK]; int id[KK];
#pragma unroll
  for (int k = 0; k < KK; ++k) { d[k] = 3.4e38f; id[k] = 0; }
  float es = 0.0f;
  const float coef = -1.0f / (2.0f * 0.1f * 0.1f);
  for (int t0 = 0; t0 < NN; t0 += 256) {
    tx[threadIdx.x] = xb[t0 + threadIdx.x];
    ty[threadIdx.x] = xb[NN + t0 + threadIdx.x];
    tz[threadIdx.x] = xb[2 * NN + t0 + threadIdx.x];
    __syncthreads();
    for (int j = 0; j < 256; ++j) {
      float dx = xi - tx[j], dy = yi - ty[j], dz = zi - tz[j];
      float dist = dx * dx + dy * dy + dz * dz;
      es += __expf(dist * coef);
      if (dist < d[KK - 1]) {
        float cur = dist; int ci = t0 + j;
#pragma unroll
        for (int k = 0; k < KK; ++k) {
          bool sw = cur < d[k];
          float td = d[k]; int ti = id[k];
          d[k] = sw ? cur : td; id[k] = sw ? ci : ti;
          cur = sw ? td : cur; ci = sw ? ti : ci;
        }
      }
    }
    __syncthreads();
  }
  long p = (long)b * NN + row;
  invd[p] = (0.25f * (float)NN) / es;   // 1/dens
#pragma unroll
  for (int k = 0; k < KK; ++k) {
    gidx[p * KK + k] = id[k];
    float nx = xb[id[k]] - xi, ny = xb[NN + id[k]] - yi, nz = xb[2 * NN + id[k]] - zi;
    long gb = (p * KK + k) * 3;
    gxyzn[gb] = nx; gxyzn[gb + 1] = ny; gxyzn[gb + 2] = nz;
  }
}

// ============================================================
// 5) dscale_pre[b,0,k,n] = invd[gidx[k]] / max_k
// ============================================================
__global__ void k_dscale(const int* __restrict__ gidx, const float* __restrict__ invd,
                         float* __restrict__ dpre) {
  long p = (long)blockIdx.x * 256 + threadIdx.x;
  if (p >= BN_TOT) return;
  int b = (int)(p / NN), n = (int)(p % NN);
  float v[KK]; float mx = 0.0f;
#pragma unroll
  for (int k = 0; k < KK; ++k) {
    v[k] = invd[(long)b * NN + gidx[p * KK + k]];
    mx = fmaxf(mx, v[k]);
  }
  float inv = 1.0f / mx;
#pragma unroll
  for (int k = 0; k < KK; ++k)
    dpre[((long)b * KK + k) * NN + n] = v[k] * inv;
}

// ============================================================
// 6) generic 1x1 conv over [B,Ci,K,N] -> [B,Co,K,N]  (small Ci/Co)
// ============================================================
template <int CI, int CO>
__global__ void k_conv_small(const float* __restrict__ in, const float* __restrict__ w,
                             const float* __restrict__ bias, float* __restrict__ out) {
  long p = (long)blockIdx.x * 256 + threadIdx.x;
  long P = (long)BB * KN;
  if (p >= P) return;
  int b = (int)(p / KN);
  long r = p % KN;
  float ac[CO];
#pragma unroll
  for (int o = 0; o < CO; ++o) ac[o] = bias[o];
#pragma unroll
  for (int c = 0; c < CI; ++c) {
    float xv = in[((long)b * CI + c) * KN + r];
#pragma unroll
    for (int o = 0; o < CO; ++o) ac[o] += w[o * CI + c] * xv;
  }
#pragma unroll
  for (int o = 0; o < CO; ++o) out[((long)b * CO + o) * KN + r] = ac[o];
}

// ============================================================
// 7) BatchNorm training stats: one workgroup per channel (deterministic)
//    x layout [Bc][C][E]; stats[c] = {mean, var}
// ============================================================
__global__ void k_bn_stats(const float* __restrict__ x, int Bc, int C, int E,
                           float* __restrict__ stats) {
  int c = blockIdx.x;
  __shared__ float s1[256], s2[256];
  float sum = 0.0f, sq = 0.0f;
  for (int b = 0; b < Bc; ++b) {
    const float* p = x + ((long)b * C + c) * E;
    for (int i = threadIdx.x; i < E; i += 256) { float v = p[i]; sum += v; sq += v * v; }
  }
  s1[threadIdx.x] = sum; s2[threadIdx.x] = sq;
  __syncthreads();
  for (int s = 128; s > 0; s >>= 1) {
    if ((int)threadIdx.x < s) { s1[threadIdx.x] += s1[threadIdx.x + s]; s2[threadIdx.x] += s2[threadIdx.x + s]; }
    __syncthreads();
  }
  if (threadIdx.x == 0) {
    float cnt = (float)Bc * (float)E;
    float m = s1[0] / cnt;
    float v = s2[0] / cnt - m * m;
    stats[c * 2] = m; stats[c * 2 + 1] = v;
  }
}

__global__ void k_bn_apply_relu(float* __restrict__ x, int C, int E, long total,
                                const float* __restrict__ stats,
                                const float* __restrict__ g, const float* __restrict__ bt) {
  long e = (long)blockIdx.x * 256 + threadIdx.x;
  if (e >= total) return;
  int c = (int)((e / E) % C);
  float m = stats[c * 2], v = stats[c * 2 + 1];
  float y = g[c] * (x[e] - m) * rsqrtf(v + EPS_BN) + bt[c];
  x[e] = y > 0.0f ? y : 0.0f;
}

// final BN+relu with scatter: in [256][BN_TOT] -> d_out [B][256][N]
__global__ void k_bn_apply_relu_out(const float* __restrict__ x, const float* __restrict__ stats,
                                    const float* __restrict__ g, const float* __restrict__ bt,
                                    float* __restrict__ out) {
  long e = (long)blockIdx.x * 256 + threadIdx.x;
  long total = 256L * BN_TOT;
  if (e >= total) return;
  int o = (int)(e / BN_TOT);
  int bn = (int)(e % BN_TOT);
  int b = bn / NN, n = bn % NN;
  float m = stats[o * 2], v = stats[o * 2 + 1];
  float y = g[o] * (x[e] - m) * rsqrtf(v + EPS_BN) + bt[o];
  out[((long)b * 256 + o) * NN + n] = y > 0.0f ? y : 0.0f;
}

// ============================================================
// 8) repack gxyz_norm [B,N,K,3] -> weightnet input [B,3,K,N]
// ============================================================
__global__ void k_pack_wn_in(const float* __restrict__ gxyzn, float* __restrict__ wnin) {
  long e = (long)blockIdx.x * 256 + threadIdx.x;
  long total = (long)BB * 3 * KN;
  if (e >= total) return;
  int b = (int)(e / (3L * KN));
  int rem = (int)(e % (3L * KN));
  int c = rem / KN;
  int r = rem % KN;
  int k = r / NN, n = r % NN;
  wnin[e] = gxyzn[(((long)b * NN + n) * KK + k) * 3 + c];
}

// ============================================================
// 9) stage A: per point, newp(259x16) @ weights^T(16x16) -> agg (bf16)
//    one 256-thread workgroup per point; LDS-staged gathered features
// ============================================================
__global__ void k_agg(const float* __restrict__ interp, const int* __restrict__ gidx,
                      const float* __restrict__ gxyzn, const float* __restrict__ dscale,
                      const float* __restrict__ wts, unsigned short* __restrict__ aggbf) {
  int p = blockIdx.x;              // point (b*N+n)
  int b = p / NN, n = p % NN;
  int t = threadIdx.x;
  __shared__ float np[KK][264];    // [k][c], padded
  __shared__ float wl[KK][KK];     // [h][k]
  __shared__ float dsc[KK];
  __shared__ int gid[KK];
  if (t < KK) {
    dsc[t] = dscale[((long)b * KK + t) * NN + n];
    gid[t] = gidx[(long)p * KK + t];
  }
  { int h = t >> 4, k = t & 15;
    wl[h][k] = wts[(((long)b * KK + h) * KK + k) * NN + n]; }
  __syncthreads();
#pragma unroll 4
  for (int k = 0; k < KK; ++k) {
    const float* row = interp + ((long)b * NN + gid[k]) * DD2;
    np[k][3 + t] = row[t] * dsc[k];
  }
  if (t < 48) {
    int k = t / 3, c = t % 3;
    np[k][c] = gxyzn[((long)p * KK + k) * 3 + c] * dsc[k];
  }
  __syncthreads();
  unsigned short* outr = aggbf + (long)p * CKP;
  for (int idx = t; idx < CK; idx += 256) {
    int c = idx >> 4, h = idx & 15;
    float s = 0.0f;
#pragma unroll
    for (int k = 0; k < KK; ++k) s += np[k][c] * wl[h][k];
    outr[idx] = f2bf(s);
  }
  if (t < CKP - CK) outr[CK + t] = 0;   // zero K padding
}

// ============================================================
// 10) weight conversions to bf16
// ============================================================
__global__ void k_cvt_w0(const float* __restrict__ w0, unsigned short* __restrict__ w0bf) {
  long e = (long)blockIdx.x * 256 + threadIdx.x;
  long total = 256L * CKP;
  if (e >= total) return;
  int o = (int)(e / CKP), ck = (int)(e % CKP);
  float v = (ck < CK) ? w0[(long)o * CK + ck] : 0.0f;
  w0bf[e] = f2bf(v);
}

__global__ void k_cvt_w1(const float* __restrict__ w1, unsigned short* __restrict__ w1bf) {
  long e = (long)blockIdx.x * 256 + threadIdx.x;
  long total = 256L * 384;
  if (e >= total) return;
  w1bf[e] = f2bf(w1[e]);
}

// ============================================================
// 11) WMMA GEMM:  C[M][Ncols] = A[M][Ktot] * B[Ncols][Ktot]^T + bias
//     A row-major bf16 (lda), Bst stored row-per-column (ldb), f32 accum.
//     Each wave computes a 64x16 strip: 4 m-tiles with 4 INDEPENDENT
//     accumulators -> dependency-free back-to-back v_wmma issue and 4x
//     reuse of each B fragment (B stream is the only one that misses L2).
// ============================================================
__global__ void k_gemm_bf16(const unsigned short* __restrict__ A, int lda,
                            const unsigned short* __restrict__ Bst, int ldb,
                            int Ktot, const float* __restrict__ bias,
                            float* __restrict__ C, int ldc, int Ncols) {
  int lane = threadIdx.x & 31;
  int wave = threadIdx.x >> 5;
  int ntile = blockIdx.x * 8 + wave;
  int mg = blockIdx.y;                  // group of 4 m-tiles (64 rows)
  int n0 = ntile * 16, m0 = mg * 64;
  if (n0 >= Ncols) return;
  int half = lane >> 4, l15 = lane & 15;
  const unsigned short* Ar = A + (long)(m0 + l15) * lda;     // A: row m = m0 + lane%16 (+16*a)
  const unsigned short* Brow = Bst + (long)(n0 + l15) * ldb; // B: column n = n0 + lane%16
  const long as16 = 16L * lda;
  v8f acc0 = {}, acc1 = {}, acc2 = {}, acc3 = {};
  for (int k0 = 0; k0 < Ktot; k0 += 32) {
    __builtin_prefetch(Brow + k0 + 256, 0, 1);   // global_prefetch_b8 on the B stream
    BF16x16 a0, a1, a2, a3, bb;
#pragma unroll
    for (int v = 0; v < 8; ++v) {
      // ISA 7.12.2: 16-bit A 16x32 — lane half selects K+8 / VGPR>=4 selects K+16
      int ka = ((v & 3) * 2) + half * 8 + (v >> 2) * 16;
      a0.u[v] = *(const unsigned int*)(Ar + k0 + ka);
      a1.u[v] = *(const unsigned int*)(Ar + as16 + k0 + ka);
      a2.u[v] = *(const unsigned int*)(Ar + 2 * as16 + k0 + ka);
      a3.u[v] = *(const unsigned int*)(Ar + 3 * as16 + k0 + ka);
      // B 32x16 — same K order as A: lane half selects K+8 / VGPR>=4 selects K+16
      int kb = ((v & 3) * 2) + half * 8 + (v >> 2) * 16;
      bb.u[v] = *(const unsigned int*)(Brow + k0 + kb);
    }
    acc0 = __builtin_amdgcn_wmma_f32_16x16x32_bf16(false, a0.b, false, bb.b, (short)0, acc0, false, false);
    acc1 = __builtin_amdgcn_wmma_f32_16x16x32_bf16(false, a1.b, false, bb.b, (short)0, acc1, false, false);
    acc2 = __builtin_amdgcn_wmma_f32_16x16x32_bf16(false, a2.b, false, bb.b, (short)0, acc2, false, false);
    acc3 = __builtin_amdgcn_wmma_f32_16x16x32_bf16(false, a3.b, false, bb.b, (short)0, acc3, false, false);
  }
  // C/D layout: lane L holds column n = L%16; VGPR r holds M = r + 8*(L/16)
#pragma unroll
  for (int r = 0; r < 8; ++r) {
    int mr = r + 8 * half;
    int col = n0 + l15;
    { int m = m0 + mr;      C[(long)m * ldc + col] = acc0[r] + bias[m]; }
    { int m = m0 + 16 + mr; C[(long)m * ldc + col] = acc1[r] + bias[m]; }
    { int m = m0 + 32 + mr; C[(long)m * ldc + col] = acc2[r] + bias[m]; }
    { int m = m0 + 48 + mr; C[(long)m * ldc + col] = acc3[r] + bias[m]; }
  }
}

// ============================================================
// 12) pack concat([relu(bn(out0)), points1]) -> xcat bf16 [BN_TOT][384]
// ============================================================
__global__ void k_pack_xcat(const float* __restrict__ x0, const float* __restrict__ points1,
                            unsigned short* __restrict__ xcat) {
  long e = (long)blockIdx.x * 256 + threadIdx.x;
  long total = (long)BN_TOT * 384;
  if (e >= total) return;
  int bn = (int)(e / 384), c = (int)(e % 384);
  float v;
  if (c < 256) {
    v = x0[(long)c * BN_TOT + bn];
  } else {
    int b = bn / NN, n = bn % NN;
    v = points1[((long)b * DD1 + (c - 256)) * NN + n];
  }
  xcat[e] = f2bf(v);
}

// ============================================================
// host-side launcher
// ============================================================
extern "C" void kernel_launch(void* const* d_in, const int* in_sizes, int n_in,
                              void* d_out, int out_size, void* d_ws, size_t ws_size,
                              hipStream_t stream) {
  (void)in_sizes; (void)n_in; (void)out_size; (void)ws_size;
  int i = 0;
  const float* xyz1    = (const float*)d_in[i++];
  const float* xyz2    = (const float*)d_in[i++];
  const float* points1 = (const float*)d_in[i++];
  const float* points2 = (const float*)d_in[i++];
  const float *dn_w[3], *dn_cb[3], *dn_g[3], *dn_b[3];
  for (int l = 0; l < 3; ++l) {
    dn_w[l] = (const float*)d_in[i++]; dn_cb[l] = (const float*)d_in[i++];
    dn_g[l] = (const float*)d_in[i++]; dn_b[l]  = (const float*)d_in[i++];
  }
  const float *wn_w[3], *wn_cb[3], *wn_g[3], *wn_b[3];
  for (int l = 0; l < 3; ++l) {
    wn_w[l] = (const float*)d_in[i++]; wn_cb[l] = (const float*)d_in[i++];
    wn_g[l] = (const float*)d_in[i++]; wn_b[l]  = (const float*)d_in[i++];
  }
  const float* w0  = (const float*)d_in[i++];
  const float* cb0 = (const float*)d_in[i++];
  const float* g0  = (const float*)d_in[i++];
  const float* b0  = (const float*)d_in[i++];
  const float* w1  = (const float*)d_in[i++];
  const float* cb1 = (const float*)d_in[i++];
  const float* g1  = (const float*)d_in[i++];
  const float* b1  = (const float*)d_in[i++];
  float* outp = (float*)d_out;

  // ---- workspace sub-allocation (256B aligned) ----
  char* ws = (char*)d_ws;
  size_t off = 0;
  auto alloc = [&](size_t bytes) -> char* {
    char* p = ws + off;
    off = (off + bytes + 255) & ~(size_t)255;
    return p;
  };
  float* p2t    = (float*)alloc((size_t)BB * SS * DD2 * 4);
  int*   idx3   = (int*)  alloc((size_t)BN_TOT * 3 * 4);
  float* w3     = (float*)alloc((size_t)BN_TOT * 3 * 4);
  float* interp = (float*)alloc((size_t)BN_TOT * DD2 * 4);
  int*   gidx   = (int*)  alloc((size_t)BN_TOT * KK * 4);
  float* invd   = (float*)alloc((size_t)BN_TOT * 4);
  float* gxyzn  = (float*)alloc((size_t)BN_TOT * KK * 3 * 4);
  float* dpre   = (float*)alloc((size_t)BB * KN * 4);
  float* mlpA   = (float*)alloc((size_t)BB * 16 * KN * 4);
  float* mlpB   = (float*)alloc((size_t)BB * 16 * KN * 4);
  float* dscout = (float*)alloc((size_t)BB * KN * 4);
  float* wnin   = (float*)alloc((size_t)BB * 3 * KN * 4);
  float* stats  = (float*)alloc((size_t)256 * 2 * 4);
  unsigned short* w0bf  = (unsigned short*)alloc((size_t)256 * CKP * 2);
  unsigned short* w1bf  = (unsigned short*)alloc((size_t)256 * 384 * 2);
  unsigned short* aggbf = (unsigned short*)alloc((size_t)BN_TOT * CKP * 2);
  unsigned short* xcat  = (unsigned short*)alloc((size_t)BN_TOT * 384 * 2);
  float* out0   = (float*)alloc((size_t)256 * BN_TOT * 4);
  float* out1   = (float*)alloc((size_t)256 * BN_TOT * 4);

  auto cdiv = [](long a, long b) { return (unsigned)((a + b - 1) / b); };

  // ---- 3-NN interpolation of points2 onto xyz1 ----
  k_transpose_p2<<<cdiv((long)BB * SS * DD2, 256), 256, 0, stream>>>(points2, p2t);
  k_3nn<<<BB * (NN / 256), 256, 0, stream>>>(xyz1, xyz2, idx3, w3);
  k_interp<<<BN_TOT, 256, 0, stream>>>(p2t, idx3, w3, interp);

  // ---- kNN + density ----
  k_knn_density<<<BB * (NN / 256), 256, 0, stream>>>(xyz1, gidx, invd, gxyzn);
  k_dscale<<<cdiv(BN_TOT, 256), 256, 0, stream>>>(gidx, invd, dpre);

  long P = (long)BB * KN;
  auto bn_layer = [&](float* x, int C, const float* g, const float* bt) {
    k_bn_stats<<<C, 256, 0, stream>>>(x, BB, C, KN, stats);
    long tot = (long)BB * C * KN;
    k_bn_apply_relu<<<cdiv(tot, 256), 256, 0, stream>>>(x, C, KN, tot, stats, g, bt);
  };

  // ---- densitynet: 1 -> 16 -> 8 -> 1 ----
  k_conv_small<1, 16><<<cdiv(P, 256), 256, 0, stream>>>(dpre, dn_w[0], dn_cb[0], mlpA);
  bn_layer(mlpA, 16, dn_g[0], dn_b[0]);
  k_conv_small<16, 8><<<cdiv(P, 256), 256, 0, stream>>>(mlpA, dn_w[1], dn_cb[1], mlpB);
  bn_layer(mlpB, 8, dn_g[1], dn_b[1]);
  k_conv_small<8, 1><<<cdiv(P, 256), 256, 0, stream>>>(mlpB, dn_w[2], dn_cb[2], dscout);
  bn_layer(dscout, 1, dn_g[2], dn_b[2]);

  // ---- weightnet: 3 -> 8 -> 8 -> 16 ----
  k_pack_wn_in<<<cdiv((long)BB * 3 * KN, 256), 256, 0, stream>>>(gxyzn, wnin);
  k_conv_small<3, 8><<<cdiv(P, 256), 256, 0, stream>>>(wnin, wn_w[0], wn_cb[0], mlpA);
  bn_layer(mlpA, 8, wn_g[0], wn_b[0]);
  k_conv_small<8, 8><<<cdiv(P, 256), 256, 0, stream>>>(mlpA, wn_w[1], wn_cb[1], mlpB);
  bn_layer(mlpB, 8, wn_g[1], wn_b[1]);
  k_conv_small<8, 16><<<cdiv(P, 256), 256, 0, stream>>>(mlpB, wn_w[2], wn_cb[2], mlpA);
  bn_layer(mlpA, 16, wn_g[2], wn_b[2]);     // mlpA now holds weights [B,16,K,N]

  // ---- stage A: per-point neighbor aggregation -> agg bf16 [BN_TOT][CKP] ----
  k_agg<<<BN_TOT, 256, 0, stream>>>(interp, gidx, gxyzn, dscout, mlpA, aggbf);

  // ---- conv0 as WMMA GEMM: [256 x 4144] x [4144 x 8192] ----
  k_cvt_w0<<<cdiv(256L * CKP, 256), 256, 0, stream>>>(w0, w0bf);
  {
    dim3 grid(BN_TOT / 128, 256 / 64);
    k_gemm_bf16<<<grid, 256, 0, stream>>>(w0bf, CKP, aggbf, CKP, CKP, cb0, out0, BN_TOT, BN_TOT);
  }
  k_bn_stats<<<256, 256, 0, stream>>>(out0, 1, 256, BN_TOT, stats);
  k_bn_apply_relu<<<cdiv(256L * BN_TOT, 256), 256, 0, stream>>>(out0, 256, BN_TOT,
                                                               256L * BN_TOT, stats, g0, b0);

  // ---- conv1 as WMMA GEMM: [256 x 384] x [384 x 8192] ----
  k_pack_xcat<<<cdiv((long)BN_TOT * 384, 256), 256, 0, stream>>>(out0, points1, xcat);
  k_cvt_w1<<<cdiv(256L * 384, 256), 256, 0, stream>>>(w1, w1bf);
  {
    dim3 grid(BN_TOT / 128, 256 / 64);
    k_gemm_bf16<<<grid, 256, 0, stream>>>(w1bf, 384, xcat, 384, 384, cb1, out1, BN_TOT, BN_TOT);
  }
  k_bn_stats<<<256, 256, 0, stream>>>(out1, 1, 256, BN_TOT, stats);
  k_bn_apply_relu_out<<<cdiv(256L * BN_TOT, 256), 256, 0, stream>>>(out1, stats, g1, b1, outp);
}
